// NNModelGraphresPerConnection_60670708023850
// MI455X (gfx1250) — matrix operand order repaired and checked by reading, not compiled, not run on hardware
//
#include <hip/hip_runtime.h>
#include <hip/hip_bf16.h>

// ---------------------------------------------------------------------------
// NNModelGraphresPerConnection pipeline for MI455X (gfx1250, wave32, WMMA).
//
// Input order assumption (pytree flatten of setup_inputs dict):
//   0 klt_data[2M,21] 1 long_data[1M,6] 2 detections[1M,3]
//   3 klt_idx[E+1] 4 long_idx[E+1] 5 detection_idx[S+1] 6 edge_idx[S+1]
//   7 edge_signs[E,1] 8 detection_signs[1M,1] 9 entry_diffs[S] 10 entry_weight[1]
//   11..24 klt_params (W,b)*7   25..38 long_params (W,b)*7
//   39..46 combine_params (W,b)*4   47..54 det_params (W,b)*4
// ---------------------------------------------------------------------------

typedef __attribute__((ext_vector_type(16))) _Float16 v16h;
typedef __attribute__((ext_vector_type(8)))  float    v8f;
typedef unsigned int u32;

union V16H { v16h v; u32 u[8]; _Float16 h[16]; };
union HU   { u32 u; _Float16 h[2]; };

struct Bias8 { const float* p[8]; };

__device__ __forceinline__ v8f wmma16(v16h a, v16h b, v8f c) {
  return __builtin_amdgcn_wmma_f32_16x16x32_f16(false, a, false, b, (short)0, c,
                                                false, false);
}

__device__ __forceinline__ v8f cinit(float b) {
  v8f c;
#pragma unroll
  for (int r = 0; r < 8; ++r) c[r] = b;
  return c;
}

// A fragment (16x32 f16 tile, M x K) from row-major LDS staging.
// Doc layout: lane L: m=L%16, half=L/16; vgpr j holds K = kbase + (j/4)*16 + half*8 + 2*(j%3... (j&3)).
__device__ __forceinline__ v16h load_A_lds(const u32* stu, int lane, int kbase,
                                           int strideU32) {
  int m = lane & 15, hf = lane >> 4;
  V16H a;
#pragma unroll
  for (int j = 0; j < 8; ++j) {
    int k = kbase + ((j >> 2) << 4) + (hf << 3) + ((j & 3) << 1);
    a.u[j] = stu[m * strideU32 + (k >> 1)];
  }
  return a.v;
}

// B fragment: 8 contiguous u32 per lane (prepacked by prep_frag_kernel).
__device__ __forceinline__ v16h load_B_frag(const u32* tile, int lane) {
  V16H b;
  const uint4* p = (const uint4*)(tile + lane * 8);
  uint4 x = p[0], y = p[1];
  b.u[0] = x.x; b.u[1] = x.y; b.u[2] = x.z; b.u[3] = x.w;
  b.u[4] = y.x; b.u[5] = y.y; b.u[6] = y.z; b.u[7] = y.w;
  return b.v;
}

// C tile (16x16 f32): reg r -> row r + 8*(lane/16), col lane%16.
__device__ __forceinline__ void store_C(_Float16* st, int strideE, int lane,
                                        int nt, v8f c, bool relu) {
  int n = lane & 15, hf = lane >> 4;
#pragma unroll
  for (int r = 0; r < 8; ++r) {
    float v = c[r];
    if (relu) v = fmaxf(v, 0.f);
    st[(hf * 8 + r) * strideE + nt * 16 + n] = (_Float16)v;
  }
}

// ---------------------------------------------------------------------------
// Weight fragment prep: f32 row-major W[K][N] -> f16 B-fragments.
// Tile t = kt*ntiles+nt; B layout: lane L: n = L%16, k = (j/4)*16 + (L/16)*8 + 2*(j%4).
// perm==1: combine layer0 row permutation (new feature order
//   [klt_n, long_n, klt_mean(64), long_mean(16), pad(14)]).
// ---------------------------------------------------------------------------
__device__ __forceinline__ float fetchW(const float* W, int k, int n, int K,
                                        int N, int perm) {
  if (n >= N) return 0.f;
  if (perm) {
    if (k >= 82) return 0.f;
    int ko = (k == 0) ? 0 : (k == 1) ? 65 : (k <= 65) ? (k - 1) : k;
    return W[(long)ko * N + n];
  }
  if (k >= K) return 0.f;
  return W[(long)k * N + n];
}

__global__ void prep_frag_kernel(const float* __restrict__ W, int Kact, int Nact,
                                 int ntiles, u32* __restrict__ dst, int perm) {
  int lane = threadIdx.x & 31;
  int t = blockIdx.x;
  int kt = t / ntiles, nt = t % ntiles;
  u32* tile = dst + (long)t * 256;
  int n = nt * 16 + (lane & 15);
  int kb = kt * 32 + ((lane >> 4) << 3);
#pragma unroll
  for (int j = 0; j < 8; ++j) {
    int k = kb + ((j >> 2) << 4) + 2 * (j & 3);
    HU x;
    x.h[0] = (_Float16)fetchW(W, k, n, Kact, Nact, perm);
    x.h[1] = (_Float16)fetchW(W, k + 1, n, Kact, Nact, perm);
    tile[lane * 8 + j] = x.u;
  }
}

// ---------------------------------------------------------------------------
// klt MLP: [N,21] -> 7 ReLU layers of width 64. Out: f16 [N][64].
// 8 waves/WG, each wave owns one 16-row tile (52 WMMAs).
// ---------------------------------------------------------------------------
__global__ __launch_bounds__(256) void klt_mlp_kernel(
    const float* __restrict__ x, int nrows, const u32* __restrict__ frags,
    Bias8 bs, u32* __restrict__ out) {
  __shared__ _Float16 stag[8][16 * 64];  // 16 KB
  const int lane = threadIdx.x & 31;
  const int wave = threadIdx.x >> 5;
  _Float16* st = stag[wave];
  u32* stu = (u32*)st;  // row stride 32 u32
  const long row0 = ((long)blockIdx.x * 8 + wave) * 16;

  __builtin_prefetch(&x[(row0 + 128) * 21], 0, 1);

  // Stage input: 16 rows x 32 cols (21 real, rest zero).
  for (int i = lane; i < 16 * 32; i += 32) {
    int r = i >> 5, c = i & 31;
    long gr = row0 + r;
    float v = (c < 21 && gr < nrows) ? x[gr * 21 + c] : 0.f;
    st[r * 64 + c] = (_Float16)v;
  }
  __syncthreads();

  // Layer 0: K=32(pad 21), N=64.
  {
    v16h a = load_A_lds(stu, lane, 0, 32);
    v8f c[4];
#pragma unroll
    for (int nt = 0; nt < 4; ++nt) {
      v8f cc = cinit(bs.p[0][nt * 16 + (lane & 15)]);
      c[nt] = wmma16(a, load_B_frag(frags + nt * 256, lane), cc);
    }
#pragma unroll
    for (int nt = 0; nt < 4; ++nt) store_C(st, 64, lane, nt, c[nt], true);
  }
  __syncthreads();

  // Layers 1..6: K=64 (2 k-chunks), N=64.
  for (int l = 1; l < 7; ++l) {
    const u32* fl = frags + (4 + (l - 1) * 8) * 256;
    v16h a0 = load_A_lds(stu, lane, 0, 32);
    v16h a1 = load_A_lds(stu, lane, 32, 32);
    v8f c[4];
#pragma unroll
    for (int nt = 0; nt < 4; ++nt) {
      v8f cc = cinit(bs.p[l][nt * 16 + (lane & 15)]);
      cc = wmma16(a0, load_B_frag(fl + nt * 256, lane), cc);
      cc = wmma16(a1, load_B_frag(fl + (4 + nt) * 256, lane), cc);
      c[nt] = cc;
    }
    __syncthreads();
#pragma unroll
    for (int nt = 0; nt < 4; ++nt) store_C(st, 64, lane, nt, c[nt], true);
    __syncthreads();
  }

  // Write scores (f16, row-major) to workspace.
  for (int i = lane; i < 16 * 32; i += 32) {
    int r = i >> 5, c = i & 31;
    long gr = row0 + r;
    if (gr < nrows) out[gr * 32 + c] = stu[r * 32 + c];
  }
}

// ---------------------------------------------------------------------------
// long MLP: [N,6] -> 7 ReLU layers of width 16. Out: f16 [N][16].
// ---------------------------------------------------------------------------
__global__ __launch_bounds__(256) void long_mlp_kernel(
    const float* __restrict__ x, int nrows, const u32* __restrict__ frags,
    Bias8 bs, u32* __restrict__ out) {
  __shared__ _Float16 stag[8][16 * 32];  // 8 KB
  const int lane = threadIdx.x & 31;
  const int wave = threadIdx.x >> 5;
  _Float16* st = stag[wave];
  u32* stu = (u32*)st;  // row stride 16 u32
  const long row0 = ((long)blockIdx.x * 8 + wave) * 16;

  // Stage input: cols 0..31, only 0..5 real; cols 16..31 stay zero forever
  // (store_C only writes cols 0..15) so hidden-layer K-padding is clean.
  for (int i = lane; i < 16 * 32; i += 32) {
    int r = i >> 5, c = i & 31;
    long gr = row0 + r;
    float v = (c < 6 && gr < nrows) ? x[gr * 6 + c] : 0.f;
    st[r * 32 + c] = (_Float16)v;
  }
  __syncthreads();

  for (int l = 0; l < 7; ++l) {
    v16h a = load_A_lds(stu, lane, 0, 16);
    v8f c = cinit(bs.p[l][lane & 15]);
    c = wmma16(a, load_B_frag(frags + l * 256, lane), c);
    __syncthreads();
    store_C(st, 32, lane, 0, c, true);
    __syncthreads();
  }

  for (int i = lane; i < 16 * 8; i += 32) {
    int r = i >> 3, c = i & 7;
    long gr = row0 + r;
    if (gr < nrows) out[gr * 8 + c] = stu[r * 16 + c];
  }
}

// ---------------------------------------------------------------------------
// det MLP: [N,3] -> 3 ReLU layers width 32, then 32->1 dot, * detection_signs.
// ---------------------------------------------------------------------------
__global__ __launch_bounds__(256) void det_mlp_kernel(
    const float* __restrict__ x, int nrows, const u32* __restrict__ frags,
    Bias8 bs, const float* __restrict__ W3, const float* __restrict__ dsign,
    float* __restrict__ detval) {
  __shared__ _Float16 stag[8][16 * 32];
  __shared__ float w3s[32];
  const int lane = threadIdx.x & 31;
  const int wave = threadIdx.x >> 5;
  _Float16* st = stag[wave];
  u32* stu = (u32*)st;
  const long row0 = ((long)blockIdx.x * 8 + wave) * 16;

  if (threadIdx.x < 32) w3s[threadIdx.x] = W3[threadIdx.x];

  for (int i = lane; i < 16 * 32; i += 32) {
    int r = i >> 5, c = i & 31;
    long gr = row0 + r;
    float v = (c < 3 && gr < nrows) ? x[gr * 3 + c] : 0.f;
    st[r * 32 + c] = (_Float16)v;
  }
  __syncthreads();

  for (int l = 0; l < 3; ++l) {
    v16h a = load_A_lds(stu, lane, 0, 16);
    v8f c[2];
#pragma unroll
    for (int nt = 0; nt < 2; ++nt) {
      v8f cc = cinit(bs.p[l][nt * 16 + (lane & 15)]);
      c[nt] = wmma16(a, load_B_frag(frags + (l * 2 + nt) * 256, lane), cc);
    }
    __syncthreads();
#pragma unroll
    for (int nt = 0; nt < 2; ++nt) store_C(st, 32, lane, nt, c[nt], true);
    __syncthreads();
  }

  // Final 32->1 dot: lanes l and l+16 each cover 16 cols of row l%16.
  int row = lane & 15, hf = lane >> 4;
  float s = 0.f;
#pragma unroll
  for (int c = 0; c < 16; ++c) s += (float)st[row * 32 + hf * 16 + c] * w3s[hf * 16 + c];
  s += __shfl_xor(s, 16, 32);
  if (lane < 16) {
    long gr = row0 + lane;
    if (gr < nrows) detval[gr] = (s + bs.p[3][0]) * dsign[gr];
  }
}

// ---------------------------------------------------------------------------
// Edge features: per-edge mean of klt scores (64) & long scores (16) + counts,
// packed as f16 [E][96] in order [klt_n, long_n, klt_mean, long_mean, pad].
// One wave per edge; rows are tiny (avg ~4).
// ---------------------------------------------------------------------------
__global__ __launch_bounds__(256) void edge_feat_kernel(
    const u32* __restrict__ klt_sc, const u32* __restrict__ long_sc,
    const int* __restrict__ klt_idx, const int* __restrict__ long_idx,
    u32* __restrict__ feat, int E) {
  int lane = threadIdx.x & 31;
  long e = (long)blockIdx.x * 8 + (threadIdx.x >> 5);
  if (e >= E) return;

  int k0 = klt_idx[e], k1 = klt_idx[e + 1];
  float sx = 0.f, sy = 0.f;
  for (int r = k0; r < k1; ++r) {
    HU w; w.u = klt_sc[(long)r * 32 + lane];
    sx += (float)w.h[0]; sy += (float)w.h[1];
  }
  float nk = (float)(k1 - k0);
  float rk = (nk > 0.f) ? 1.f / nk : 0.f;
  sx *= rk; sy *= rk;

  int l0 = long_idx[e], l1 = long_idx[e + 1];
  float lx = 0.f, ly = 0.f;
  if (lane < 8) {
    for (int r = l0; r < l1; ++r) {
      HU w; w.u = long_sc[(long)r * 8 + lane];
      lx += (float)w.h[0]; ly += (float)w.h[1];
    }
  }
  float nl = (float)(l1 - l0);
  float rl = (nl > 0.f) ? 1.f / nl : 0.f;
  lx *= rl; ly *= rl;

  u32* fe = feat + (long)e * 48;
  HU kp; kp.h[0] = (_Float16)sx; kp.h[1] = (_Float16)sy;
  fe[1 + lane] = kp.u;                 // klt mean pairs: u32 1..32
  if (lane < 15) {
    HU lp; lp.h[0] = (_Float16)lx; lp.h[1] = (_Float16)ly;
    fe[33 + lane] = (lane < 8) ? lp.u : 0u;  // long mean 33..40, pad 41..47
  }
  if (lane == 0) {
    HU np; np.h[0] = (_Float16)nk; np.h[1] = (_Float16)nl;
    fe[0] = np.u;                      // counts
  }
}

// ---------------------------------------------------------------------------
// combine MLP: f16 [E][96] -> 256 -> 256 -> 256 (ReLU) -> 1, * edge_signs.
// 4 waves/WG; per wave tile: 48 + 128 + 128 WMMAs + dot.
// ---------------------------------------------------------------------------
__device__ __forceinline__ void combine_layer(const u32* stu, _Float16* st,
                                              int lane, const u32* fl,
                                              const float* bias, int ktiles) {
  v8f c[16];
#pragma unroll
  for (int nt = 0; nt < 16; ++nt) c[nt] = cinit(bias[nt * 16 + (lane & 15)]);
  for (int kt = 0; kt < ktiles; ++kt) {
    v16h a = load_A_lds(stu, lane, kt * 32, 128);
    const u32* fk = fl + (long)kt * 16 * 256;
#pragma unroll
    for (int nt = 0; nt < 16; ++nt)
      c[nt] = wmma16(a, load_B_frag(fk + nt * 256, lane), c[nt]);
  }
  __syncthreads();
#pragma unroll
  for (int nt = 0; nt < 16; ++nt) store_C(st, 256, lane, nt, c[nt], true);
  __syncthreads();
}

__global__ __launch_bounds__(128) void combine_mlp_kernel(
    const u32* __restrict__ feat, int E, const u32* __restrict__ frags,
    Bias8 bs, const float* __restrict__ W3, const float* __restrict__ esign,
    float* __restrict__ conn) {
  __shared__ _Float16 stag[4][16 * 256];  // 32 KB
  __shared__ float w3s[256];
  const int lane = threadIdx.x & 31;
  const int wave = threadIdx.x >> 5;
  _Float16* st = stag[wave];
  u32* stu = (u32*)st;  // row stride 128 u32
  const long row0 = ((long)blockIdx.x * 4 + wave) * 16;

  for (int i = threadIdx.x; i < 256; i += 128) w3s[i] = W3[i];

  // Stage 16 x 48 u32 of features (already f16).
  for (int i = lane; i < 16 * 48; i += 32) {
    int r = i / 48, c = i % 48;
    long gr = row0 + r;
    stu[r * 128 + c] = (gr < E) ? feat[gr * 48 + c] : 0u;
  }
  __syncthreads();

  combine_layer(stu, st, lane, frags, bs.p[0], 3);                 // 96 -> 256
  combine_layer(stu, st, lane, frags + 48 * 256, bs.p[1], 8);      // 256 -> 256
  combine_layer(stu, st, lane, frags + 176 * 256, bs.p[2], 8);     // 256 -> 256

  // Final 256->1 dot: lanes l/l+16 cover halves of row l%16.
  int row = lane & 15, hf = lane >> 4;
  float s = 0.f;
  for (int c = 0; c < 128; ++c)
    s += (float)st[row * 256 + hf * 128 + c] * w3s[hf * 128 + c];
  s += __shfl_xor(s, 16, 32);
  if (lane < 16) {
    long gr = row0 + lane;
    if (gr < E) conn[gr] = (s + bs.p[3][0]) * esign[gr];
  }
}

// ---------------------------------------------------------------------------
// Final: out[s] = sum conn over edge segment + sum detval over det segment
//                 + entry_diffs[s]*entry_weight.
// ---------------------------------------------------------------------------
__global__ __launch_bounds__(256) void final_kernel(
    const float* __restrict__ conn, const float* __restrict__ detval,
    const int* __restrict__ edge_idx, const int* __restrict__ det_idx,
    const float* __restrict__ entry_diffs, const float* __restrict__ entry_w,
    float* __restrict__ out, int S) {
  int s = blockIdx.x * 256 + threadIdx.x;
  if (s >= S) return;
  float acc = entry_diffs[s] * entry_w[0];
  int e0 = edge_idx[s], e1 = edge_idx[s + 1];
  for (int i = e0; i < e1; ++i) acc += conn[i];
  int d0 = det_idx[s], d1 = det_idx[s + 1];
  for (int i = d0; i < d1; ++i) acc += detval[i];
  out[s] = acc;
}

// ---------------------------------------------------------------------------
// Host launcher.
// ---------------------------------------------------------------------------
extern "C" void kernel_launch(void* const* d_in, const int* in_sizes, int n_in,
                              void* d_out, int out_size, void* d_ws,
                              size_t ws_size, hipStream_t stream) {
  (void)ws_size; (void)n_in;
  const int N_KLT  = in_sizes[0] / 21;
  const int N_LONG = in_sizes[1] / 6;
  const int N_DET  = in_sizes[2] / 3;
  const int E      = in_sizes[3] - 1;
  const int S      = out_size;

  // ---- workspace carve (all regions fully written before read) ----
  char* ws = (char*)d_ws;
  u32* frags = (u32*)ws;                                  // 369 KB of fragments
  size_t off = 512 * 1024;
  u32* klt_sc = (u32*)(ws + off); off += (size_t)N_KLT * 64 * 2;
  off = (off + 255) & ~(size_t)255;
  u32* long_sc = (u32*)(ws + off); off += (size_t)N_LONG * 16 * 2;
  off = (off + 255) & ~(size_t)255;
  u32* feat = (u32*)(ws + off); off += (size_t)E * 96 * 2;
  off = (off + 255) & ~(size_t)255;
  float* conn = (float*)(ws + off); off += (size_t)E * 4;
  off = (off + 255) & ~(size_t)255;
  float* detval = (float*)(ws + off);

  // ---- param pointers ----
  Bias8 kb{}, lb{}, cb{}, db{};
  for (int l = 0; l < 7; ++l) kb.p[l] = (const float*)d_in[12 + 2 * l];
  for (int l = 0; l < 7; ++l) lb.p[l] = (const float*)d_in[26 + 2 * l];
  for (int l = 0; l < 4; ++l) cb.p[l] = (const float*)d_in[40 + 2 * l];
  for (int l = 0; l < 4; ++l) db.p[l] = (const float*)d_in[48 + 2 * l];

  // ---- weight fragment prep ----
  // klt: tiles [0..51]
  prep_frag_kernel<<<4, 32, 0, stream>>>((const float*)d_in[11], 21, 64, 4,
                                         frags + 0 * 256, 0);
  for (int l = 1; l < 7; ++l)
    prep_frag_kernel<<<8, 32, 0, stream>>>((const float*)d_in[11 + 2 * l], 64,
                                           64, 4, frags + (4 + (l - 1) * 8) * 256, 0);
  // long: tiles [52..58]
  for (int l = 0; l < 7; ++l)
    prep_frag_kernel<<<1, 32, 0, stream>>>((const float*)d_in[25 + 2 * l],
                                           l == 0 ? 6 : 16, 16, 1,
                                           frags + (52 + l) * 256, 0);
  // det: tiles [59..64]
  for (int l = 0; l < 3; ++l)
    prep_frag_kernel<<<2, 32, 0, stream>>>((const float*)d_in[47 + 2 * l],
                                           l == 0 ? 3 : 32, 32, 2,
                                           frags + (59 + 2 * l) * 256, 0);
  // combine: tiles [65..368]; layer0 uses permuted feature rows
  prep_frag_kernel<<<48, 32, 0, stream>>>((const float*)d_in[39], 82, 256, 16,
                                          frags + 65 * 256, 1);
  prep_frag_kernel<<<128, 32, 0, stream>>>((const float*)d_in[41], 256, 256, 16,
                                           frags + 113 * 256, 0);
  prep_frag_kernel<<<128, 32, 0, stream>>>((const float*)d_in[43], 256, 256, 16,
                                           frags + 241 * 256, 0);

  // ---- MLPs ----
  klt_mlp_kernel<<<(N_KLT + 127) / 128, 256, 0, stream>>>(
      (const float*)d_in[0], N_KLT, frags, kb, klt_sc);
  long_mlp_kernel<<<(N_LONG + 127) / 128, 256, 0, stream>>>(
      (const float*)d_in[1], N_LONG, frags + 52 * 256, lb, long_sc);
  det_mlp_kernel<<<(N_DET + 127) / 128, 256, 0, stream>>>(
      (const float*)d_in[2], N_DET, frags + 59 * 256, db,
      (const float*)d_in[53], (const float*)d_in[8], detval);

  // ---- segment means -> features ----
  edge_feat_kernel<<<(E + 7) / 8, 256, 0, stream>>>(
      klt_sc, long_sc, (const int*)d_in[3], (const int*)d_in[4], feat, E);

  // ---- combine MLP -> signed connection scores ----
  combine_mlp_kernel<<<(E + 63) / 64, 128, 0, stream>>>(
      feat, E, frags + 65 * 256, cb, (const float*)d_in[45],
      (const float*)d_in[7], conn);

  // ---- final segment sums ----
  final_kernel<<<(S + 255) / 256, 256, 0, stream>>>(
      conn, detval, (const int*)d_in[6], (const int*)d_in[5],
      (const float*)d_in[9], (const float*)d_in[10], (float*)d_out, S);
}
